// GPT_43533788513030
// MI455X (gfx1250) — matrix operand order repaired and checked by reading, not compiled, not run on hardware
//
#include <hip/hip_runtime.h>
#include <hip/hip_bf16.h>
#include <math.h>

// ---------------------------------------------------------------------------
// GPT forward for gfx1250 (CDNA5, wave32, WMMA + async global->LDS staging).
// B=2, T=2048, D=1024, H=16, DH=64, L=6, V=50257, TGT=512 (keys < 1537 always visible)
// ---------------------------------------------------------------------------

typedef __attribute__((ext_vector_type(16))) _Float16 v16h;
typedef __attribute__((ext_vector_type(8)))  _Float16 v8h;
typedef __attribute__((ext_vector_type(8)))  float    v8f;
typedef __attribute__((ext_vector_type(4)))  float    v4f;

#define GPT_B   2
#define GPT_T   2048
#define GPT_D   1024
#define GPT_H   16
#define GPT_DH  64
#define GPT_L   6
#define GPT_V   50257
#define GPT_M   (GPT_B * GPT_T)     // 4096 tokens
#define MASK_FREE 1537              // keys < T-TGT+1 always attended

// ---- Async global->LDS (CDNA5 GLOBAL_LOAD_ASYNC_TO_LDS_B128, ASYNCcnt) ----
// Builtin signature (probed via diagnostics): arg0 = AS(1) int4* global src,
// arg1 = LDS int4* dest, then imm offset, imm cpol. Generic->AS casts done
// via integer round-trip (low 32 bits of a generic LDS pointer are the
// wave-relative LDS offset per the aperture rules).

#if defined(__has_builtin)
#if __has_builtin(__builtin_amdgcn_global_load_async_to_lds_b128)
#define HAVE_ASYNC_COPY 1
#endif
#endif
#ifndef HAVE_ASYNC_COPY
#define HAVE_ASYNC_COPY 0
#endif

typedef int v4i __attribute__((vector_size(16)));
typedef __attribute__((address_space(1))) v4i* as1_v4i;
typedef __attribute__((address_space(3))) v4i* as3_v4i;

__device__ __forceinline__ void wait_async0() {
#if defined(__has_builtin) && __has_builtin(__builtin_amdgcn_s_wait_asynccnt)
  __builtin_amdgcn_s_wait_asynccnt(0);
#else
  asm volatile("s_wait_asynccnt 0x0" ::: "memory");
#endif
}

__device__ __forceinline__ void copy16B_to_lds(const _Float16* g,
                                               _Float16* lds) {
#if HAVE_ASYNC_COPY
  __builtin_amdgcn_global_load_async_to_lds_b128(
      (as1_v4i)(unsigned long long)(g),
      (as3_v4i)(unsigned int)(unsigned long long)(lds), 0, 0);
#else
  *(v8h*)lds = *(const v8h*)g;
#endif
}

// ---- WMMA helpers ---------------------------------------------------------

__device__ __forceinline__ v8f wmma_f32_f16(v16h a, v16h b, v8f c) {
  // (neg_a, A, neg_b, B, c_mod, C, reuse_a, reuse_b)
  return __builtin_amdgcn_wmma_f32_16x16x32_f16(false, a, false, b, (short)0, c,
                                                false, false);
}

// A-operand (16x32 f16) per-lane fragment, ISA layout:
//   lanes 0-15 : K = 0..7  and 16..23   (kh=0)
//   lanes 16-31: K = 8..15 and 24..31   (kh=1)
// p must point at &row[kh*8]; loads p[0..7] and p[16..23].
__device__ __forceinline__ v16h frag_interleaved(const _Float16* p) {
  v8h lo = *(const v8h*)(p);
  v8h hi = *(const v8h*)(p + 16);
  return __builtin_shufflevector(lo, hi, 0, 1, 2, 3, 4, 5, 6, 7,
                                 8, 9, 10, 11, 12, 13, 14, 15);
}

// B-operand (32x16 f16) per-lane fragment (column n = lane&15):
//   lanes 0-15 : K = 0..7 and 16..23 ; lanes 16-31: K = 8..15 and 24..31
// p must point at &col_row[kh*8]; loads p[0..7] and p[16..23].
__device__ __forceinline__ v16h frag_contig(const _Float16* p) {
  v8h lo = *(const v8h*)(p);
  v8h hi = *(const v8h*)(p + 16);
  return __builtin_shufflevector(lo, hi, 0, 1, 2, 3, 4, 5, 6, 7,
                                 8, 9, 10, 11, 12, 13, 14, 15);
}

// ---- Embedding gather -----------------------------------------------------

__global__ __launch_bounds__(256) void embed_kernel(
    const int* __restrict__ idx, const float* __restrict__ emb,
    float* __restrict__ X) {
  const int token = blockIdx.x;
  const int t = idx[token];
  const float* src = emb + (size_t)t * GPT_D;
  float* dst = X + (size_t)token * GPT_D;
  for (int i = threadIdx.x; i < GPT_D; i += 256) dst[i] = src[i];
}

// ---- LayerNorm (fp32 in -> f16 out) ---------------------------------------

__global__ __launch_bounds__(256) void ln_kernel(
    const float* __restrict__ X, const float* __restrict__ W,
    const float* __restrict__ Bv, _Float16* __restrict__ Out) {
  const int row = blockIdx.x;
  const int tid = threadIdx.x;
  const float* x = X + (size_t)row * GPT_D;
  float s = 0.f, s2 = 0.f;
  for (int i = tid; i < GPT_D; i += 256) {
    float v = x[i];
    s += v;
    s2 += v * v;
  }
  __shared__ float red[256];
  __shared__ float red2[256];
  red[tid] = s;
  red2[tid] = s2;
  __syncthreads();
  for (int off = 128; off > 0; off >>= 1) {
    if (tid < off) {
      red[tid] += red[tid + off];
      red2[tid] += red2[tid + off];
    }
    __syncthreads();
  }
  const float mean = red[0] * (1.0f / GPT_D);
  const float var = red2[0] * (1.0f / GPT_D) - mean * mean;
  const float rstd = rsqrtf(var + 1e-5f);
  _Float16* o = Out + (size_t)row * GPT_D;
  for (int i = tid; i < GPT_D; i += 256) {
    float v = (x[i] - mean) * rstd * W[i] + Bv[i];
    o[i] = (_Float16)v;
  }
}

// ---- Tiled WMMA GEMM:  C[M,N] = A[M,K](f16) @ B[K,N](f32) + bias ----------
// Block tile 128x64, BK=64 (two WMMA K-steps per LDS stage), 256 threads =
// 8 waves in a 4x2 wave grid, each wave computing 32x32 via 2x2 WMMA tiles.
// A tile staged via async global->LDS (ASYNCcnt); B tile converted fp32->f16
// and transposed by the VALU (DMA cannot convert).
// grid.x = M tiles (fast-moving) so concurrently-dispatched blocks share the
// same B column stripe -> B weight stripes are read from HBM once and hit in
// the 192MB L2 for sibling blocks (critical for the 206MB LM-head weights).
// EPI: 0 = f16 out, 1 = f16 out with exact GELU, 2 = f32 out += residual,
//      3 = f32 out (no bias if bias==nullptr).

template <int EPI>
__global__ __launch_bounds__(256) void gemm_kernel(
    const _Float16* __restrict__ A, const float* __restrict__ Bw,
    const float* __restrict__ bias, float* __restrict__ out_f32,
    _Float16* __restrict__ out_f16, const float* __restrict__ resid,
    int M, int N, int K) {
  constexpr int BK = 64;
  __shared__ _Float16 As[128][72];  // row stride 72 halves = 144B (16B mult.)
  __shared__ _Float16 Bs[64][72];   // stored transposed: Bs[n][k]

  const int tid = threadIdx.x;
  const int lane = tid & 31;
  const int wave = tid >> 5;
  const int wm = wave >> 1;  // 0..3
  const int wn = wave & 1;   // 0..1
  const int row0 = blockIdx.x * 128;   // M tile (fast grid axis)
  const int col0 = blockIdx.y * 64;    // N tile (slow grid axis)
  const int kh = lane >> 4;
  const int l15 = lane & 15;
  const bool full_n = (col0 + 64 <= N);

  v8f acc[2][2] = {};

  for (int k0 = 0; k0 < K; k0 += BK) {
    // Stage A tile (f16): 128 rows x 64 halves; 4 async b128 per thread.
    {
      const int m = tid >> 1;
      const int kk = (tid & 1) * 32;
      const _Float16* g = A + (size_t)(row0 + m) * K + k0 + kk;
      _Float16* d = &As[m][kk];
#pragma unroll
      for (int j = 0; j < 4; ++j) copy16B_to_lds(g + j * 8, d + j * 8);
    }
    // Stage B tile: 64 k x 64 n fp32 -> f16, transposed into Bs[n][k].
    // 16 consecutive floats per thread (coalesced), vector fast path.
    {
      const int kk = tid >> 2;           // 0..63
      const int n0 = (tid & 3) * 16;     // 0,16,32,48
      const float* g = Bw + (size_t)(k0 + kk) * N + col0 + n0;
      if (full_n) {
#pragma unroll
        for (int j4 = 0; j4 < 4; ++j4) {
          v4f v = *(const v4f*)(g + j4 * 4);
#pragma unroll
          for (int j = 0; j < 4; ++j)
            Bs[n0 + j4 * 4 + j][kk] = (_Float16)v[j];
        }
      } else {
#pragma unroll
        for (int j = 0; j < 16; ++j) {
          float v = (col0 + n0 + j < N) ? g[j] : 0.f;
          Bs[n0 + j][kk] = (_Float16)v;
        }
      }
    }
#if HAVE_ASYNC_COPY
    wait_async0();
#endif
    __syncthreads();

#pragma unroll
    for (int ks = 0; ks < 64; ks += 32) {
      v16h af[2], bf[2];
#pragma unroll
      for (int tm = 0; tm < 2; ++tm)
        af[tm] = frag_interleaved(&As[wm * 32 + tm * 16 + l15][ks + kh * 8]);
#pragma unroll
      for (int tn = 0; tn < 2; ++tn)
        bf[tn] = frag_contig(&Bs[wn * 32 + tn * 16 + l15][ks + kh * 8]);

#pragma unroll
      for (int tm = 0; tm < 2; ++tm)
#pragma unroll
        for (int tn = 0; tn < 2; ++tn)
          acc[tm][tn] = wmma_f32_f16(af[tm], bf[tn], acc[tm][tn]);
    }
    __syncthreads();
  }

  // Epilogue. C layout: VGPR r holds row r (lanes 0-15) / row r+8 (lanes
  // 16-31), column = lane&15.
#pragma unroll
  for (int tm = 0; tm < 2; ++tm) {
#pragma unroll
    for (int tn = 0; tn < 2; ++tn) {
      const int col = col0 + wn * 32 + tn * 16 + l15;
      if (col >= N) continue;
      const float bval = bias ? bias[col] : 0.f;
#pragma unroll
      for (int r = 0; r < 8; ++r) {
        const int row = row0 + wm * 32 + tm * 16 + kh * 8 + r;
        float v = acc[tm][tn][r] + bval;
        const size_t o = (size_t)row * N + col;
        if constexpr (EPI == 0) {
          out_f16[o] = (_Float16)v;
        } else if constexpr (EPI == 1) {
          float g = 0.5f * v * (1.0f + erff(v * 0.70710678118654752f));
          out_f16[o] = (_Float16)g;
        } else if constexpr (EPI == 2) {
          out_f32[o] = resid[o] + v;
        } else {
          out_f32[o] = v;
        }
      }
    }
  }
}

// ---- Flash attention (WMMA for QK^T and PV, online softmax) ---------------
// grid = (T/64, H, B); 128 threads = 4 waves; wave w owns 16 query rows.

__global__ __launch_bounds__(128) void attn_kernel(
    const _Float16* __restrict__ Q, const _Float16* __restrict__ Kb,
    const _Float16* __restrict__ Vb, _Float16* __restrict__ Y) {
  const int qt = blockIdx.x;   // query tile (64 rows)
  const int hd = blockIdx.y;   // head
  const int bb = blockIdx.z;   // batch
  const int tid = threadIdx.x;
  const int lane = tid & 31;
  const int wave = tid >> 5;
  const int kh = lane >> 4;
  const int l15 = lane & 15;

  __shared__ _Float16 Ks[64][72];       // [key][d]  (B layout for QK^T)
  __shared__ _Float16 Vt[64][72];       // [d][key]  (B layout for PV)
  __shared__ _Float16 Ps[4][16][72];    // per-wave P staging (C->A transpose)

  const int qrow0 = qt * 64 + wave * 16;
  const size_t base = ((size_t)bb * GPT_T) * GPT_D + (size_t)hd * GPT_DH;

  // Q fragments for both K=32 windows of DH=64, loaded once.
  v16h qf[2];
  {
    const _Float16* qp = Q + base + (size_t)(qrow0 + l15) * GPT_D;
    qf[0] = frag_interleaved(qp + kh * 8);
    qf[1] = frag_interleaved(qp + 32 + kh * 8);
  }

  v8f y[4] = {};
  float mrow[8], lrow[8];
#pragma unroll
  for (int r = 0; r < 8; ++r) {
    mrow[r] = -__builtin_inff();
    lrow[r] = 0.f;
  }

  const int qt_end = qt * 64 + 63;
  const int kb_max = (qt_end > (MASK_FREE - 1)) ? qt_end : (MASK_FREE - 1);

  for (int kb = 0; kb <= kb_max; kb += 64) {
    // Stage K tile (async DMA to LDS) and transposed V tile (manual).
#pragma unroll
    for (int i = 0; i < 4; ++i) {
      const int e = tid + i * 128;       // 0..511
      const int key = e >> 3;            // 0..63
      const int d8 = (e & 7) * 8;        // 0..56
      const size_t off = base + (size_t)(kb + key) * GPT_D + d8;
      copy16B_to_lds(Kb + off, &Ks[key][d8]);
      v8h vv = *(const v8h*)(Vb + off);
#pragma unroll
      for (int j = 0; j < 8; ++j) Vt[d8 + j][key] = vv[j];
    }
#if HAVE_ASYNC_COPY
    wait_async0();
#endif
    __syncthreads();

    // S = (Q K^T) * scale, 4 tiles of 16 keys each.
    v8f s[4];
#pragma unroll
    for (int tn = 0; tn < 4; ++tn) {
      v8f a = {};
      v16h b0 = frag_contig(&Ks[tn * 16 + l15][kh * 8]);
      v16h b1 = frag_contig(&Ks[tn * 16 + l15][32 + kh * 8]);
      a = wmma_f32_f16(qf[0], b0, a);
      a = wmma_f32_f16(qf[1], b1, a);
      s[tn] = a;
    }

    // Scale + mask + per-row local max.
    float mloc[8];
#pragma unroll
    for (int r = 0; r < 8; ++r) {
      const int row = qt * 64 + wave * 16 + kh * 8 + r;
      float mx = -__builtin_inff();
#pragma unroll
      for (int tn = 0; tn < 4; ++tn) {
        const int key = kb + tn * 16 + l15;
        float v = s[tn][r] * 0.125f;  // 1/sqrt(64)
        const bool ok = (key < MASK_FREE) || (key <= row);
        v = ok ? v : -__builtin_inff();
        s[tn][r] = v;
        mx = fmaxf(mx, v);
      }
      mloc[r] = mx;
    }
    // Butterfly max within each 16-lane half (lanes share the same 8 rows).
#pragma unroll
    for (int r = 0; r < 8; ++r) {
      float m = mloc[r];
      m = fmaxf(m, __shfl_xor(m, 1, 32));
      m = fmaxf(m, __shfl_xor(m, 2, 32));
      m = fmaxf(m, __shfl_xor(m, 4, 32));
      m = fmaxf(m, __shfl_xor(m, 8, 32));
      mloc[r] = m;
    }
    float corr[8];
#pragma unroll
    for (int r = 0; r < 8; ++r) {
      const float mnew = fmaxf(mrow[r], mloc[r]);
      corr[r] = expf(mrow[r] - mnew);  // 0 on first tile (mrow = -inf)
      mrow[r] = mnew;
    }
    // P = exp(S - m), row sums.
#pragma unroll
    for (int r = 0; r < 8; ++r) {
      float sum = 0.f;
#pragma unroll
      for (int tn = 0; tn < 4; ++tn) {
        float p = expf(s[tn][r] - mrow[r]);  // masked (-inf) -> 0
        s[tn][r] = p;
        sum += p;
      }
      sum += __shfl_xor(sum, 1, 32);
      sum += __shfl_xor(sum, 2, 32);
      sum += __shfl_xor(sum, 4, 32);
      sum += __shfl_xor(sum, 8, 32);
      lrow[r] = lrow[r] * corr[r] + sum;
    }
    // Rescale accumulator.
#pragma unroll
    for (int tn = 0; tn < 4; ++tn)
#pragma unroll
      for (int r = 0; r < 8; ++r) y[tn][r] *= corr[r];

    // C-layout -> A-layout via per-wave LDS staging (LDS ops are in-order
    // within a wave, no block barrier needed for this buffer).
#pragma unroll
    for (int tn = 0; tn < 4; ++tn)
#pragma unroll
      for (int r = 0; r < 8; ++r)
        Ps[wave][kh * 8 + r][tn * 16 + l15] = (_Float16)s[tn][r];

    v16h pf0 = frag_interleaved(&Ps[wave][l15][kh * 8]);
    v16h pf1 = frag_interleaved(&Ps[wave][l15][32 + kh * 8]);

    // Y += P @ V (contract over 64 keys = two K=32 windows).
#pragma unroll
    for (int tn = 0; tn < 4; ++tn) {
      v16h v0 = frag_contig(&Vt[tn * 16 + l15][kh * 8]);
      v16h v1 = frag_contig(&Vt[tn * 16 + l15][32 + kh * 8]);
      y[tn] = wmma_f32_f16(pf0, v0, y[tn]);
      y[tn] = wmma_f32_f16(pf1, v1, y[tn]);
    }
    __syncthreads();  // protect Ks/Vt before next tile's staging
  }

  // Normalize and write.
#pragma unroll
  for (int tn = 0; tn < 4; ++tn)
#pragma unroll
    for (int r = 0; r < 8; ++r) {
      const int row = qrow0 + kh * 8 + r;
      const float v = y[tn][r] / lrow[r];
      Y[base + (size_t)row * GPT_D + tn * 16 + l15] = (_Float16)v;
    }
}

// ---------------------------------------------------------------------------

extern "C" void kernel_launch(void* const* d_in, const int* in_sizes, int n_in,
                              void* d_out, int out_size, void* d_ws,
                              size_t ws_size, hipStream_t stream) {
  (void)in_sizes; (void)n_in; (void)out_size; (void)ws_size;

  const int* idx       = (const int*)d_in[0];
  const float* tok_emb = (const float*)d_in[1];
  const float* ln1_w   = (const float*)d_in[2];
  const float* ln1_b   = (const float*)d_in[3];
  const float* Wq      = (const float*)d_in[4];
  const float* bq      = (const float*)d_in[5];
  const float* Wk      = (const float*)d_in[6];
  const float* bk      = (const float*)d_in[7];
  const float* Wv      = (const float*)d_in[8];
  const float* bv      = (const float*)d_in[9];
  const float* Wp      = (const float*)d_in[10];
  const float* bp      = (const float*)d_in[11];
  const float* ln2_w   = (const float*)d_in[12];
  const float* ln2_b   = (const float*)d_in[13];
  const float* W1      = (const float*)d_in[14];
  const float* b1      = (const float*)d_in[15];
  const float* W2      = (const float*)d_in[16];
  const float* b2      = (const float*)d_in[17];
  const float* lnf_w   = (const float*)d_in[18];
  const float* lnf_b   = (const float*)d_in[19];
  const float* head_w  = (const float*)d_in[20];

  const int M = GPT_M, D = GPT_D, V = GPT_V;

  // Workspace carve-up (needs ~92.3 MB).
  char* ws = (char*)d_ws;
  float*     x   = (float*)ws;                 ws += (size_t)M * D * 4;  // 16 MB
  _Float16*  h   = (_Float16*)ws;              ws += (size_t)M * D * 2;  //  8 MB
  _Float16*  qb  = (_Float16*)ws;              ws += (size_t)M * D * 2;
  _Float16*  kb  = (_Float16*)ws;              ws += (size_t)M * D * 2;
  _Float16*  vb  = (_Float16*)ws;              ws += (size_t)M * D * 2;
  _Float16*  yb  = (_Float16*)ws;              ws += (size_t)M * D * 2;
  _Float16*  hid = (_Float16*)ws;              ws += (size_t)M * 4 * D * 2;  // 32 MB

  dim3 blk(256);
  // grid.x = M tiles (fast axis) for L2-friendly B-stripe reuse.
  dim3 gD(M / 128, D / 64);            // N=1024 GEMMs
  dim3 g4D(M / 128, 4 * D / 64);       // N=4096 GEMM
  dim3 gV(M / 128, (V + 63) / 64);     // head GEMM (786 column stripes)
  dim3 gAttn(GPT_T / 64, GPT_H, GPT_B);

  embed_kernel<<<M, blk, 0, stream>>>(idx, tok_emb, x);

  for (int l = 0; l < GPT_L; ++l) {
    const size_t oDD = (size_t)l * D * D;
    const size_t oD  = (size_t)l * D;
    const size_t oD4 = (size_t)l * D * 4 * D;

    ln_kernel<<<M, blk, 0, stream>>>(x, ln1_w + oD, ln1_b + oD, h);

    gemm_kernel<0><<<gD, blk, 0, stream>>>(h, Wq + oDD, bq + oD, nullptr, qb,
                                           nullptr, M, D, D);
    gemm_kernel<0><<<gD, blk, 0, stream>>>(h, Wk + oDD, bk + oD, nullptr, kb,
                                           nullptr, M, D, D);
    gemm_kernel<0><<<gD, blk, 0, stream>>>(h, Wv + oDD, bv + oD, nullptr, vb,
                                           nullptr, M, D, D);

    attn_kernel<<<gAttn, dim3(128), 0, stream>>>(qb, kb, vb, yb);

    // x = x + y @ Wp + bp
    gemm_kernel<2><<<gD, blk, 0, stream>>>(yb, Wp + oDD, bp + oD, x, nullptr, x,
                                           M, D, D);

    ln_kernel<<<M, blk, 0, stream>>>(x, ln2_w + oD, ln2_b + oD, h);

    // hid = gelu(h @ W1 + b1)
    gemm_kernel<1><<<g4D, blk, 0, stream>>>(h, W1 + oD4, b1 + (size_t)l * 4 * D,
                                            nullptr, hid, nullptr, M, 4 * D, D);
    // x = x + hid @ W2 + b2
    gemm_kernel<2><<<gD, blk, 0, stream>>>(hid, W2 + oD4, b2 + oD, x, nullptr,
                                           x, M, D, 4 * D);
  }

  ln_kernel<<<M, blk, 0, stream>>>(x, lnf_w, lnf_b, h);

  // logits = h @ head_w  (no bias), fp32 straight to d_out
  gemm_kernel<3><<<gV, blk, 0, stream>>>(h, head_w, nullptr, (float*)d_out,
                                         nullptr, nullptr, M, V, D);
}
